// TransformerModel_86792699118246
// MI455X (gfx1250) — matrix operand order repaired and checked by reading, not compiled, not run on hardware
//
#include <hip/hip_runtime.h>

#define TOK   4096
#define SEQ   2048
#define DIM   1024
#define D3    3072
#define MLPD  4096
#define NLAYER 6

typedef __attribute__((ext_vector_type(16))) __bf16 v16bf;
typedef __attribute__((ext_vector_type(8)))  float  v8f;
typedef __attribute__((ext_vector_type(4)))  unsigned int v4u;
typedef __attribute__((ext_vector_type(4)))  float  v4f;

union FragAB { v16bf v; __bf16 e[16]; v4u q[2]; };
union V32    { v4u q[4]; __bf16 e[32]; };

// float -> bf16 round-to-nearest-even via integer math (no reliance on native cvt)
static __device__ __forceinline__ __bf16 f2bf(float f) {
  union { float f; unsigned u; } a; a.f = f;
  unsigned r = a.u + 0x7FFFu + ((a.u >> 16) & 1u);
  union { unsigned short s; __bf16 b; } c; c.s = (unsigned short)(r >> 16);
  return c.b;
}

static __device__ __forceinline__ v8f wmma_bf16(v16bf a, v16bf b, v8f c) {
  // D(16x16,f32) = A(16x32,bf16) * B(32x16,bf16) + C
  return __builtin_amdgcn_wmma_f32_16x16x32_bf16(false, a, false, b, (short)0, c,
                                                 false, false);
}

// ---------------------------------------------------------------------------
// gfx1250 async global->LDS copy (ASYNCcnt path), with safe fallback.
// Builtin prototype (from toolchain diagnostic): param0 = int4(vector_size)
// pointer in AS1 (global/"__device__"), param1 = int4 pointer in AS3 (LDS),
// then imm offset + imm cpol.
// ---------------------------------------------------------------------------
#if defined(__AMDGCN__) && __has_builtin(__builtin_amdgcn_global_load_async_to_lds_b128)
#define HAVE_ASYNC_LDS 1
typedef int v4i_vs __attribute__((vector_size(16)));
typedef __attribute__((address_space(1))) v4i_vs* as1_v4i;
typedef __attribute__((address_space(3))) v4i_vs* as3_v4i;
#endif

static __device__ __forceinline__ void async_copy_b128(const __bf16* g, __bf16* l) {
#ifdef HAVE_ASYNC_LDS
  __builtin_amdgcn_global_load_async_to_lds_b128((as1_v4i)g, (as3_v4i)l, 0, 0);
#else
  *(v4u*)l = *(const v4u*)g;
#endif
}

static __device__ __forceinline__ void async_wait0() {
#ifdef HAVE_ASYNC_LDS
#if __has_builtin(__builtin_amdgcn_s_wait_asynccnt)
  __builtin_amdgcn_s_wait_asynccnt(0);
#else
  asm volatile("s_wait_asynccnt 0x0" ::: "memory");
#endif
#endif
}

// ---------------------------------------------------------------------------
// LayerNorm: one block per token (row of 1024), fp32 in, bf16 out
// ---------------------------------------------------------------------------
__global__ __launch_bounds__(256) void ln_kernel(const float* __restrict__ x,
                                                 const float* __restrict__ g,
                                                 const float* __restrict__ bta,
                                                 __bf16* __restrict__ y) {
  __shared__ float red[256];
  __shared__ float stat[2];
  const int tok = blockIdx.x, tid = threadIdx.x;
  const float* row = x + (size_t)tok * DIM;

  float v[4];
  float s = 0.f;
#pragma unroll
  for (int i = 0; i < 4; ++i) { v[i] = row[tid + i * 256]; s += v[i]; }
  red[tid] = s; __syncthreads();
  for (int o = 128; o > 0; o >>= 1) {
    if (tid < o) red[tid] += red[tid + o];
    __syncthreads();
  }
  if (tid == 0) stat[0] = red[0] * (1.0f / DIM);
  __syncthreads();
  const float m = stat[0];

  float q = 0.f;
#pragma unroll
  for (int i = 0; i < 4; ++i) { float d = v[i] - m; q += d * d; }
  red[tid] = q; __syncthreads();
  for (int o = 128; o > 0; o >>= 1) {
    if (tid < o) red[tid] += red[tid + o];
    __syncthreads();
  }
  if (tid == 0) stat[1] = rsqrtf(red[0] * (1.0f / DIM) + 1e-5f);
  __syncthreads();
  const float rs = stat[1];

#pragma unroll
  for (int i = 0; i < 4; ++i) {
    int idx = tid + i * 256;
    y[(size_t)tok * DIM + idx] = f2bf((v[i] - m) * rs * g[idx] + bta[idx]);
  }
}

// ---------------------------------------------------------------------------
// GEMM: C[M,N] = A(bf16,[M,K]) @ B(f32,[K,N]) (+bias), fused epilogues.
// MODE 0: store bf16   MODE 1: exact-GELU then bf16   MODE 2: f32 += (residual)
// Block: 256 thr = 8 waves, tile 128x128; wave tile 64x32; K-step 32.
// A tile staged via global_load_async_to_lds_b128 (ASYNCcnt) when available;
// B tile converted f32->bf16 + transposed through VGPRs (overlaps the async copy).
// ---------------------------------------------------------------------------
template <int MODE>
__global__ __launch_bounds__(256) void gemm_bf16_kernel(
    const __bf16* __restrict__ A, const float* __restrict__ B,
    const float* __restrict__ bias, void* __restrict__ Cout,
    int M, int N, int K) {
  __shared__ __bf16 lA[128 * 32];   // [row][k]
  __shared__ __bf16 lB[128 * 32];   // [col][k]  (transposed during staging)

  const int tid  = threadIdx.x;
  const int wid  = tid >> 5, lane = tid & 31;
  const int hl   = lane >> 4, lidx = lane & 15;
  const int wM   = wid >> 2, wN = wid & 3;          // waves: 2 x 4
  const int bN   = blockIdx.x * 128, bM = blockIdx.y * 128;

  const v8f vzero = {0.f, 0.f, 0.f, 0.f, 0.f, 0.f, 0.f, 0.f};
  v8f acc[4][2];
#pragma unroll
  for (int i = 0; i < 4; ++i)
#pragma unroll
    for (int j = 0; j < 2; ++j) acc[i][j] = vzero;

  const int arow = tid >> 1, aseg = (tid & 1) * 16;   // A staging map
  const int bkr  = tid >> 3, bch  = (tid & 7) * 16;   // B staging map

  for (int k0 = 0; k0 < K; k0 += 32) {
    __syncthreads();
    {  // A tile: 128 rows x 32 k -- async DMA into LDS (2 x b128 per thread)
      const __bf16* src = A + (size_t)(bM + arow) * K + k0 + aseg;
      async_copy_b128(src,     &lA[arow * 32 + aseg]);
      async_copy_b128(src + 8, &lA[arow * 32 + aseg + 8]);
    }
    {  // B tile: 32 k x 128 cols f32 -> bf16, transposed to [col][k]
      const float* src = B + (size_t)(k0 + bkr) * N + bN + bch;
#pragma unroll
      for (int c = 0; c < 16; c += 4) {
        v4f f = *(const v4f*)(src + c);
        lB[(bch + c + 0) * 32 + bkr] = f2bf(f.x);
        lB[(bch + c + 1) * 32 + bkr] = f2bf(f.y);
        lB[(bch + c + 2) * 32 + bkr] = f2bf(f.z);
        lB[(bch + c + 3) * 32 + bkr] = f2bf(f.w);
      }
    }
    if (k0 + 32 < K) {  // global_prefetch_b8 on next K-tile
      __builtin_prefetch(A + (size_t)(bM + arow) * K + k0 + 32 + aseg, 0, 0);
      __builtin_prefetch(B + (size_t)(k0 + 32 + bkr) * N + bN + bch, 0, 0);
    }
    async_wait0();      // wave-local: all async LDS writes landed
    __syncthreads();    // workgroup: tiles visible to every wave

    FragAB bF[2];
#pragma unroll
    for (int ns = 0; ns < 2; ++ns) {  // B frag: k = (e>>3)*16 + hl*8 + (e&7)
      const __bf16* p = &lB[(wN * 32 + ns * 16 + lidx) * 32];
      bF[ns].q[0] = *(const v4u*)(p + hl * 8);
      bF[ns].q[1] = *(const v4u*)(p + 16 + hl * 8);
    }
#pragma unroll
    for (int ms = 0; ms < 4; ++ms) {  // A frag: k = (e>>3)*16 + hl*8 + (e&7)
      FragAB aF;
      const __bf16* p = &lA[(wM * 64 + ms * 16 + lidx) * 32];
      aF.q[0] = *(const v4u*)(p + hl * 8);
      aF.q[1] = *(const v4u*)(p + 16 + hl * 8);
#pragma unroll
      for (int ns = 0; ns < 2; ++ns)
        acc[ms][ns] = wmma_bf16(aF.v, bF[ns].v, acc[ms][ns]);
    }
  }

  // Epilogue. C layout: row m = r + 8*hl, col n = lidx.
#pragma unroll
  for (int ms = 0; ms < 4; ++ms) {
#pragma unroll
    for (int ns = 0; ns < 2; ++ns) {
      const int gcol = bN + wN * 32 + ns * 16 + lidx;
      const float bv = bias ? bias[gcol] : 0.0f;
#pragma unroll
      for (int r = 0; r < 8; ++r) {
        const int grow = bM + wM * 64 + ms * 16 + r + 8 * hl;
        float v = acc[ms][ns][r] + bv;
        if (MODE == 0) {
          ((__bf16*)Cout)[(size_t)grow * N + gcol] = f2bf(v);
        } else if (MODE == 1) {
          v = 0.5f * v * (1.0f + erff(v * 0.70710678118f));
          ((__bf16*)Cout)[(size_t)grow * N + gcol] = f2bf(v);
        } else {
          float* O = (float*)Cout;
          O[(size_t)grow * N + gcol] += v;
        }
      }
    }
  }
}

// ---------------------------------------------------------------------------
// Flash attention. qkv: bf16 [b, n, 3, H=16, 64]. o: bf16 [b*n, 1024].
// Grid (32 q-tiles, 16 heads, 2 batch), block 128 = 4 waves x 16 q-rows.
// ---------------------------------------------------------------------------
__global__ __launch_bounds__(128) void attn_kernel(const __bf16* __restrict__ qkv,
                                                   __bf16* __restrict__ o) {
  __shared__ __bf16 Vt[64 * 64];      // [dhead][key] transposed V tile (shared)
  __shared__ __bf16 Pt[4][16 * 64];   // per-wave probabilities, [m][key]

  const int tid  = threadIdx.x;
  const int wid  = tid >> 5, lane = tid & 31;
  const int hl   = lane >> 4, lidx = lane & 15;
  const int b    = blockIdx.z, head = blockIdx.y;
  const int q0   = blockIdx.x * 64 + wid * 16;
  const __bf16* base = qkv + (size_t)b * SEQ * D3;
  const float SC = 0.03125f;  // DIM^-0.5 = 1/32 (reference scales by full dim)

  // Q fragments (constant for the whole kernel): A layout, head_dim 64 -> 2 k-steps
  FragAB qf[2];
  {
    const __bf16* qp = base + (size_t)(q0 + lidx) * D3 + head * 64;
#pragma unroll
    for (int ks = 0; ks < 2; ++ks) {
      qf[ks].q[0] = *(const v4u*)(qp + ks * 32 + hl * 8);
      qf[ks].q[1] = *(const v4u*)(qp + ks * 32 + 16 + hl * 8);
    }
  }

  const v8f vzero = {0.f, 0.f, 0.f, 0.f, 0.f, 0.f, 0.f, 0.f};
  float m_run[8], l_run[8];
  v8f oacc[4];
#pragma unroll
  for (int r = 0; r < 8; ++r) { m_run[r] = -3.0e38f; l_run[r] = 0.f; }
#pragma unroll
  for (int nt = 0; nt < 4; ++nt) oacc[nt] = vzero;

  for (int kt0 = 0; kt0 < SEQ; kt0 += 64) {
    __syncthreads();
    {  // stage V tile transposed: Vt[dh][key]
      const int key = tid >> 1, dh0 = (tid & 1) * 32;
      const __bf16* vp = base + (size_t)(kt0 + key) * D3 + 2048 + head * 64 + dh0;
      V32 vv;
      vv.q[0] = *(const v4u*)(vp);
      vv.q[1] = *(const v4u*)(vp + 8);
      vv.q[2] = *(const v4u*)(vp + 16);
      vv.q[3] = *(const v4u*)(vp + 24);
#pragma unroll
      for (int j = 0; j < 32; ++j) Vt[(dh0 + j) * 64 + key] = vv.e[j];
    }
    __syncthreads();

    // S = (Q K^T) * SC  -- 4 key sub-tiles of 16
    v8f sacc[4];
#pragma unroll
    for (int st = 0; st < 4; ++st) {
      sacc[st] = vzero;
#pragma unroll
      for (int ks = 0; ks < 2; ++ks) {
        FragAB kf;  // B frag: K-index = dhead (= ks*32 + (e>>3)*16 + hl*8 + (e&7)), N = key
        const __bf16* kp =
            base + (size_t)(kt0 + st * 16 + lidx) * D3 + 1024 + head * 64 + ks * 32;
        kf.q[0] = *(const v4u*)(kp + hl * 8);
        kf.q[1] = *(const v4u*)(kp + 16 + hl * 8);
        sacc[st] = wmma_bf16(qf[ks].v, kf.v, sacc[st]);
      }
#pragma unroll
      for (int r = 0; r < 8; ++r) sacc[st][r] *= SC;
    }

    // row max across 4 tiles then across the 16 lanes of this half
    float vm[8];
#pragma unroll
    for (int r = 0; r < 8; ++r)
      vm[r] = fmaxf(fmaxf(sacc[0][r], sacc[1][r]), fmaxf(sacc[2][r], sacc[3][r]));
#pragma unroll
    for (int r = 0; r < 8; ++r) {
      float m = vm[r];
      m = fmaxf(m, __shfl_xor(m, 1, 16));
      m = fmaxf(m, __shfl_xor(m, 2, 16));
      m = fmaxf(m, __shfl_xor(m, 4, 16));
      m = fmaxf(m, __shfl_xor(m, 8, 16));
      vm[r] = m;
    }

    // online-softmax rescale of running state + O accumulators
#pragma unroll
    for (int r = 0; r < 8; ++r) {
      float nm = fmaxf(m_run[r], vm[r]);
      float sc = __expf(m_run[r] - nm);
      m_run[r] = nm;
      l_run[r] *= sc;
#pragma unroll
      for (int nt = 0; nt < 4; ++nt) oacc[nt][r] *= sc;
    }

    // exponentiate + row sums
    float vs[8];
#pragma unroll
    for (int r = 0; r < 8; ++r) vs[r] = 0.f;
#pragma unroll
    for (int st = 0; st < 4; ++st)
#pragma unroll
      for (int r = 0; r < 8; ++r) {
        float p = __expf(sacc[st][r] - m_run[r]);
        sacc[st][r] = p;
        vs[r] += p;
      }
#pragma unroll
    for (int r = 0; r < 8; ++r) {
      float s = vs[r];
      s += __shfl_xor(s, 1, 16);
      s += __shfl_xor(s, 2, 16);
      s += __shfl_xor(s, 4, 16);
      s += __shfl_xor(s, 8, 16);
      l_run[r] += s;
    }

    // P -> LDS (row m = r + 8*hl, col = st*16 + lidx), bf16
#pragma unroll
    for (int st = 0; st < 4; ++st)
#pragma unroll
      for (int r = 0; r < 8; ++r)
        Pt[wid][(r + 8 * hl) * 64 + st * 16 + lidx] = f2bf(sacc[st][r]);

    // O += P @ V  (K-dim = 64 keys -> 2 k-steps; 4 dhead tiles of 16)
#pragma unroll
    for (int nt = 0; nt < 4; ++nt) {
#pragma unroll
      for (int ks = 0; ks < 2; ++ks) {
        FragAB pa, vf;
        const __bf16* pp = &Pt[wid][lidx * 64 + ks * 32];
        pa.q[0] = *(const v4u*)(pp + hl * 8);
        pa.q[1] = *(const v4u*)(pp + 16 + hl * 8);
        const __bf16* vb = &Vt[(nt * 16 + lidx) * 64 + ks * 32];
        vf.q[0] = *(const v4u*)(vb + hl * 8);
        vf.q[1] = *(const v4u*)(vb + 16 + hl * 8);
        oacc[nt] = wmma_bf16(pa.v, vf.v, oacc[nt]);
      }
    }
  }

  // normalize and write O (bf16, [token][head*64+dh])
#pragma unroll
  for (int nt = 0; nt < 4; ++nt)
#pragma unroll
    for (int r = 0; r < 8; ++r) {
      const int qrow = q0 + r + 8 * hl;
      const int col  = head * 64 + nt * 16 + lidx;
      o[(size_t)(b * SEQ + qrow) * DIM + col] = f2bf(oacc[nt][r] / l_run[r]);
    }
}

// ---------------------------------------------------------------------------
// Host orchestration
// ---------------------------------------------------------------------------
extern "C" void kernel_launch(void* const* d_in, const int* in_sizes, int n_in,
                              void* d_out, int out_size, void* d_ws, size_t ws_size,
                              hipStream_t stream) {
  (void)in_sizes; (void)n_in; (void)out_size; (void)ws_size;
  const float* x     = (const float*)d_in[0];
  const float* Wqkv  = (const float*)d_in[1];
  const float* Wout  = (const float*)d_in[2];
  const float* bout  = (const float*)d_in[3];
  const float* ln1_g = (const float*)d_in[4];
  const float* ln1_b = (const float*)d_in[5];
  const float* ln2_g = (const float*)d_in[6];
  const float* ln2_b = (const float*)d_in[7];
  const float* W1    = (const float*)d_in[8];
  const float* b1    = (const float*)d_in[9];
  const float* W2    = (const float*)d_in[10];
  const float* b2    = (const float*)d_in[11];

  char* ws = (char*)d_ws;
  float*  h    = (float*)(ws);                            // 16.8 MB fp32 residual
  __bf16* y    = (__bf16*)(ws + (size_t)16777216);        //  8.4 MB
  __bf16* qkvb = (__bf16*)(ws + (size_t)25165824);        // 25.2 MB
  __bf16* ob   = (__bf16*)(ws + (size_t)50331648);        //  8.4 MB
  __bf16* ub   = (__bf16*)(ws + (size_t)58720256);        // 33.6 MB

  (void)hipMemcpyAsync(h, x, (size_t)TOK * DIM * sizeof(float),
                       hipMemcpyDeviceToDevice, stream);

  const dim3 blkG(256), blkA(128), blkL(256);
  for (int l = 0; l < NLAYER; ++l) {
    const float* wqkv = Wqkv + (size_t)l * DIM * D3;
    const float* wout = Wout + (size_t)l * DIM * DIM;
    const float* bo   = bout + (size_t)l * DIM;
    const float* g1   = ln1_g + (size_t)l * DIM;
    const float* be1  = ln1_b + (size_t)l * DIM;
    const float* g2   = ln2_g + (size_t)l * DIM;
    const float* be2  = ln2_b + (size_t)l * DIM;
    const float* w1   = W1 + (size_t)l * DIM * MLPD;
    const float* bf1  = b1 + (size_t)l * MLPD;
    const float* w2   = W2 + (size_t)l * MLPD * DIM;
    const float* bf2  = b2 + (size_t)l * DIM;

    // y = LN1(h)
    ln_kernel<<<TOK, blkL, 0, stream>>>(h, g1, be1, y);
    // qkv = y @ Wqkv (no bias), bf16
    gemm_bf16_kernel<0><<<dim3(D3 / 128, TOK / 128), blkG, 0, stream>>>(
        y, wqkv, nullptr, qkvb, TOK, D3, DIM);
    // o = softmax(q k^T * scale) v
    attn_kernel<<<dim3(SEQ / 64, 16, 2), blkA, 0, stream>>>(qkvb, ob);
    // h += o @ Wout + bout
    gemm_bf16_kernel<2><<<dim3(DIM / 128, TOK / 128), blkG, 0, stream>>>(
        ob, wout, bo, h, TOK, DIM, DIM);
    // y = LN2(h)
    ln_kernel<<<TOK, blkL, 0, stream>>>(h, g2, be2, y);
    // u = gelu(y @ W1 + b1), bf16
    gemm_bf16_kernel<1><<<dim3(MLPD / 128, TOK / 128), blkG, 0, stream>>>(
        y, w1, bf1, ub, TOK, MLPD, DIM);
    // h += u @ W2 + b2
    gemm_bf16_kernel<2><<<dim3(DIM / 128, TOK / 128), blkG, 0, stream>>>(
        ub, w2, bf2, h, TOK, DIM, MLPD);
  }

  (void)hipMemcpyAsync(d_out, h, (size_t)TOK * DIM * sizeof(float),
                       hipMemcpyDeviceToDevice, stream);
}
